// MMD_loss_17257178596130
// MI455X (gfx1250) — matrix operand order repaired and checked by reading, not compiled, not run on hardware
//
#include <hip/hip_runtime.h>
#include <hip/hip_bf16.h>

typedef __attribute__((ext_vector_type(16))) _Float16 v16h;
typedef __attribute__((ext_vector_type(8)))  _Float16 v8h;
typedef __attribute__((ext_vector_type(4)))  _Float16 v4h;
typedef __attribute__((ext_vector_type(8)))  float    v8f;
typedef int v4i __attribute__((vector_size(16)));   // b128 payload type for async-to-LDS builtin

constexpr int D      = 512;    // feature dim
constexpr int HALF   = 4096;   // rows in source (== rows in target)
constexpr int NROWS  = 8192;   // total rows
constexpr int TILE   = 128;    // workgroup tile (rows and cols)
constexpr int NT     = NROWS / TILE;   // 64 tiles per dim
constexpr int LDS_STRIDE = 80; // bytes per 32-f16 row: 64B data + 16B pad (16B aligned, de-conflicted)
constexpr float LOG2E = 1.4426950408889634f;

union Frag { v16h v; v8h h[2]; };

// ---- CDNA5 async copy (ASYNCcnt path), builtin if present else inline asm
__device__ __forceinline__ void async_copy_b128(const _Float16* g, void* l) {
#if __has_builtin(__builtin_amdgcn_global_load_async_to_lds_b128)
    __builtin_amdgcn_global_load_async_to_lds_b128(
        (__attribute__((address_space(1))) v4i*)g,
        (__attribute__((address_space(3))) v4i*)l, 0, 0);
#else
    unsigned loff = (unsigned)(size_t)(__attribute__((address_space(3))) void*)l;
    asm volatile("global_load_async_to_lds_b128 %0, %1, off"
                 :: "v"(loff), "v"(g) : "memory");
#endif
}
__device__ __forceinline__ void wait_async0() {
#if __has_builtin(__builtin_amdgcn_s_wait_asynccnt)
    __builtin_amdgcn_s_wait_asynccnt(0);
#else
    asm volatile("s_wait_asynccnt 0x0" ::: "memory");
#endif
}

// -------------------------------------------- prep: f16 copy of concat(total) + row norms
__global__ __launch_bounds__(256) void prep_kernel(const float* __restrict__ src,
                                                   const float* __restrict__ tgt,
                                                   _Float16* __restrict__ Th,
                                                   float* __restrict__ sq) {
    __shared__ float sh[256];
    const int t  = threadIdx.x;
    const int rr = t >> 7;          // 2 rows / block
    const int tr = t & 127;         // 4 floats / thread
    const int row = blockIdx.x * 2 + rr;
    const float* p = (row < HALF) ? (src + (size_t)row * D)
                                  : (tgt + (size_t)(row - HALF) * D);
    float4 v = reinterpret_cast<const float4*>(p)[tr];
    union { v4h v; _Float16 e[4]; } h;
    h.e[0] = (_Float16)v.x; h.e[1] = (_Float16)v.y;
    h.e[2] = (_Float16)v.z; h.e[3] = (_Float16)v.w;
    *reinterpret_cast<v4h*>(Th + (size_t)row * D + tr * 4) = h.v;
    sh[t] = v.x * v.x + v.y * v.y + v.z * v.z + v.w * v.w;   // fp32 norms (exact path)
    __syncthreads();
    for (int s = 64; s > 0; s >>= 1) {
        if (tr < s) sh[t] += sh[t + s];
        __syncthreads();
    }
    if (tr == 0) sq[row] = sh[t];
}

// ------------------------------------------------- fused GEMM + epilogue
// PASS 1: per-tile sum of clamped d2 (for bandwidth).
// PASS 2: per-tile signed sum of  sum_q exp(-d2 / (bw * 2^q)).
template <int PASS>
__global__ __launch_bounds__(256) void mmd_pass_kernel(const _Float16* __restrict__ Th,
                                                       const float* __restrict__ sq,
                                                       const float* __restrict__ accv,
                                                       float* __restrict__ partial) {
    __shared__ alignas(16) unsigned char ldsA[2][TILE * LDS_STRIDE];
    __shared__ alignas(16) unsigned char ldsB[2][TILE * LDS_STRIDE];
    __shared__ float sh[256];

    const int bi = blockIdx.y;
    const int bj = blockIdx.x;
    if (bj < bi) {                         // symmetry: only upper-triangular tiles
        if (threadIdx.x == 0) partial[bi * NT + bj] = 0.0f;
        return;
    }
    const bool diag  = (bi == bj);
    const int  ibase = bi * TILE;
    const int  jbase = bj * TILE;
    const _Float16* __restrict__ aTh = Th + (size_t)ibase * D;
    const _Float16* __restrict__ bTh = Th + (size_t)jbase * D;

    const int tid  = threadIdx.x;
    const int wave = tid >> 5, lane = tid & 31;
    const int wr   = wave >> 1;     // 0..3 : wave-row  (32 rows each)
    const int wc   = wave & 1;      // 0..1 : wave-col  (64 cols each)
    const int l16  = lane & 15;
    const int lh   = lane >> 4;     // lane half selects K chunk

    v8f acc8[8] = {};               // 2 (row tiles) x 4 (col tiles), fp32 accum

    // staging assignment: thread -> (row, 32B half-row), two b128 chunks each panel
    const int sr   = tid >> 1;                       // 0..127
    const int soff = (tid & 1) * 32;                 // LDS byte offset in row
    const int gelt = (tid & 1) * 16;                 // f16 element offset in row
    const unsigned ldsoff = sr * LDS_STRIDE + soff;

    // prologue: issue copies for k-step 0 into buffer 0
    {
        const _Float16* ga = aTh + (size_t)sr * D + gelt;
        async_copy_b128(ga,     &ldsA[0][ldsoff]);
        async_copy_b128(ga + 8, &ldsA[0][ldsoff + 16]);
        if (!diag) {
            const _Float16* gb = bTh + (size_t)sr * D + gelt;
            async_copy_b128(gb,     &ldsB[0][ldsoff]);
            async_copy_b128(gb + 8, &ldsB[0][ldsoff + 16]);
        }
    }

    int buf = 0;
    for (int kk = 0; kk < D; kk += 32) {
        wait_async0();              // this wave's copies for `buf` are in LDS
        __syncthreads();            // ... and everyone else's too; prev compute done

        if (kk + 32 < D) {          // overlap next panel's DMA with this tile's WMMAs
            const int nb = buf ^ 1;
            const _Float16* ga = aTh + (size_t)sr * D + (kk + 32) + gelt;
            async_copy_b128(ga,     &ldsA[nb][ldsoff]);
            async_copy_b128(ga + 8, &ldsA[nb][ldsoff + 16]);
            if (!diag) {
                const _Float16* gb = bTh + (size_t)sr * D + (kk + 32) + gelt;
                async_copy_b128(gb,     &ldsB[nb][ldsoff]);
                async_copy_b128(gb + 8, &ldsB[nb][ldsoff + 16]);
            }
        }

        const unsigned char* pA = ldsA[buf];
        const unsigned char* pB = diag ? ldsA[buf] : ldsB[buf];  // X.X^T: B frag = row panel
        Frag a[2], b[4];
        // A 16x32 f16 layout: lanes 0-15 K={0..7,16..23}, lanes 16-31 K={8..15,24..31}
        #pragma unroll
        for (int ti = 0; ti < 2; ++ti) {
            const unsigned char* p = &pA[(wr * 32 + ti * 16 + l16) * LDS_STRIDE + lh * 16];
            a[ti].h[0] = *reinterpret_cast<const v8h*>(p);
            a[ti].h[1] = *reinterpret_cast<const v8h*>(p + 32);
        }
        // B 32x16 f16 layout: lanes 0-15 K={0..7,16..23}, lanes 16-31 K={8..15,24..31}
        #pragma unroll
        for (int tj = 0; tj < 4; ++tj) {
            const unsigned char* p = &pB[(wc * 64 + tj * 16 + l16) * LDS_STRIDE + lh * 16];
            b[tj].h[0] = *reinterpret_cast<const v8h*>(p);
            b[tj].h[1] = *reinterpret_cast<const v8h*>(p + 32);
        }
        #pragma unroll
        for (int ti = 0; ti < 2; ++ti)
            #pragma unroll
            for (int tj = 0; tj < 4; ++tj)
                acc8[ti * 4 + tj] = __builtin_amdgcn_wmma_f32_16x16x32_f16(
                    false, a[ti].v, false, b[tj].v, (short)0, acc8[ti * 4 + tj], false, false);

        buf ^= 1;
    }

    // ---- epilogue: d2 = sq_i + sq_j - 2g, clamp, then reduce
    float sqi[2][8], sqj[4];
    #pragma unroll
    for (int ti = 0; ti < 2; ++ti)
        #pragma unroll
        for (int e = 0; e < 8; ++e)
            sqi[ti][e] = sq[ibase + wr * 32 + ti * 16 + lh * 8 + e];
    #pragma unroll
    for (int tj = 0; tj < 4; ++tj)
        sqj[tj] = sq[jbase + wc * 64 + tj * 16 + l16];

    float local = 0.0f;
    if (PASS == 1) {
        #pragma unroll
        for (int ti = 0; ti < 2; ++ti)
            #pragma unroll
            for (int tj = 0; tj < 4; ++tj)
                #pragma unroll
                for (int e = 0; e < 8; ++e) {
                    float d2 = fmaxf(sqi[ti][e] + sqj[tj] - 2.0f * acc8[ti * 4 + tj][e], 0.0f);
                    local += d2;
                }
    } else {
        const float bw = accv[0];
        float cq[5];
        #pragma unroll
        for (int q = 0; q < 5; ++q) cq[q] = -LOG2E / (bw * (float)(1 << q));
        #pragma unroll
        for (int ti = 0; ti < 2; ++ti)
            #pragma unroll
            for (int tj = 0; tj < 4; ++tj)
                #pragma unroll
                for (int e = 0; e < 8; ++e) {
                    float d2 = fmaxf(sqi[ti][e] + sqj[tj] - 2.0f * acc8[ti * 4 + tj][e], 0.0f);
                    #pragma unroll
                    for (int q = 0; q < 5; ++q)
                        local += __builtin_amdgcn_exp2f(d2 * cq[q]);   // v_exp_f32
                }
    }

    sh[tid] = local;
    __syncthreads();
    for (int s = 128; s > 0; s >>= 1) {
        if (tid < s) sh[tid] += sh[tid + s];
        __syncthreads();
    }
    if (tid == 0) {
        float w = diag ? 1.0f : 2.0f;                  // symmetric mirror tile
        float sgn = 1.0f;
        if (PASS == 2) sgn = ((bi < NT / 2) == (bj < NT / 2)) ? 1.0f : -1.0f;
        partial[bi * NT + bj] = sh[0] * w * sgn;
    }
}

// -------------------------------------------------- small deterministic reductions
__global__ __launch_bounds__(256) void reduce_bw_kernel(const float* __restrict__ partial,
                                                        float* __restrict__ accv) {
    __shared__ float sh[256];
    float local = 0.0f;
    for (int i = threadIdx.x; i < NT * NT; i += 256) local += partial[i];
    sh[threadIdx.x] = local;
    __syncthreads();
    for (int s = 128; s > 0; s >>= 1) {
        if (threadIdx.x < s) sh[threadIdx.x] += sh[threadIdx.x + s];
        __syncthreads();
    }
    if (threadIdx.x == 0) {
        // bandwidth = sum(d2)/(n^2-n) / KERNEL_MUL^(KERNEL_NUM//2)
        accv[0] = sh[0] / 67100672.0f / 4.0f;   // 8192*8191 exact in fp32
    }
}

__global__ __launch_bounds__(256) void finalize_kernel(const float* __restrict__ partial,
                                                       float* __restrict__ out) {
    __shared__ float sh[256];
    float local = 0.0f;
    for (int i = threadIdx.x; i < NT * NT; i += 256) local += partial[i];
    sh[threadIdx.x] = local;
    __syncthreads();
    for (int s = 128; s > 0; s >>= 1) {
        if (threadIdx.x < s) sh[threadIdx.x] += sh[threadIdx.x + s];
        __syncthreads();
    }
    if (threadIdx.x == 0) out[0] = sh[0] * (1.0f / (4096.0f * 4096.0f));
}

// -------------------------------------------------------------- launcher
extern "C" void kernel_launch(void* const* d_in, const int* in_sizes, int n_in,
                              void* d_out, int out_size, void* d_ws, size_t ws_size,
                              hipStream_t stream) {
    (void)in_sizes; (void)n_in; (void)out_size; (void)ws_size;
    const float* src = (const float*)d_in[0];
    const float* tgt = (const float*)d_in[1];
    float* out = (float*)d_out;

    // workspace layout: [ f16 copy of total : 8 MiB ][ sq : 32 KB ][ partial : 16 KB ][ bw ]
    _Float16* Th  = (_Float16*)d_ws;
    float*    sqv = (float*)d_ws + (size_t)NROWS * D / 2;   // f16 copy = NROWS*D/2 floats
    float*    part = sqv + NROWS;
    float*    accv = part + NT * NT;

    prep_kernel<<<NROWS / 2, 256, 0, stream>>>(src, tgt, Th, sqv);
    mmd_pass_kernel<1><<<dim3(NT, NT), 256, 0, stream>>>(Th, sqv, accv, part);
    reduce_bw_kernel<<<1, 256, 0, stream>>>(part, accv);
    mmd_pass_kernel<2><<<dim3(NT, NT), 256, 0, stream>>>(Th, sqv, accv, part);
    finalize_kernel<<<1, 256, 0, stream>>>(part, out);
}
